// SplashAttentionWrapper_27187142983765
// MI455X (gfx1250) — compile-verified, matrix operand order checked by reading, not run on hardware
//
#include <hip/hip_runtime.h>
#include <hip/hip_bf16.h>

#define BDIM 2
#define SDIM 2048
#define DDIM 2048
#define HNUM 16
#define KVNUM 4
#define HD 128
#define CAPV 50.0f
#define SCALEV 0.08838834764831845f   // 128^-0.5

typedef __attribute__((ext_vector_type(16))) __bf16 v16bf;
typedef __attribute__((ext_vector_type(8)))  __bf16 v8bf;
typedef __attribute__((ext_vector_type(4)))  __bf16 v4bf;
typedef __attribute__((ext_vector_type(8)))  float  v8f;
typedef unsigned int u32x4 __attribute__((ext_vector_type(4)));
typedef int          i32x8 __attribute__((ext_vector_type(8)));
typedef int          i32x4 __attribute__((ext_vector_type(4)));

static __device__ __forceinline__ __bf16 f2bf(float f) {
  unsigned u = __builtin_bit_cast(unsigned, f);
  u = (u + 0x7FFFu + ((u >> 16) & 1u)) >> 16;
  unsigned short s = (unsigned short)u;
  return __builtin_bit_cast(__bf16, s);
}

// LDS byte offset of a __shared__ object (generic ptr low 32 bits = LDS offset)
static __device__ __forceinline__ unsigned lds_off(const void* p) {
  return (unsigned)(unsigned long long)p;
}

// Issue one TDM 2D tile load: global (row-major, stride0 elements/row, 2-byte
// elements) -> LDS packed tile of tile_d1 rows x tile_d0 elements.
// D# layout per CDNA5 ISA. This toolchain's
// builtin takes 6 args (g0, g1, g2, g3, g4-extra, cpol).
static __device__ __forceinline__ void tdm_load_2d(unsigned lds_byte_off,
                                                   const void* g,
                                                   unsigned tile_d0,
                                                   unsigned tile_d1,
                                                   unsigned stride0) {
  const unsigned long long ga = (unsigned long long)g;
  u32x4 g0;
  g0.x = 1u;                                                  // count=1
  g0.y = lds_byte_off;                                        // lds_addr
  g0.z = (unsigned)ga;                                        // global[31:0]
  g0.w = (unsigned)((ga >> 32) & 0x01FFFFFFu) | 0x80000000u;  // global[56:32] | type=2
  const unsigned td0 = 1u << 20, td1 = 1u << 20;              // generous OOB bounds
  i32x8 g1;
  g1[0] = (int)(1u << 16);                                    // data_size=1 (2 bytes)
  g1[1] = (int)((td0 & 0xFFFFu) << 16);                       // tensor_dim0[15:0]
  g1[2] = (int)(((td0 >> 16) & 0xFFFFu) | ((td1 & 0xFFFFu) << 16));
  g1[3] = (int)(((td1 >> 16) & 0xFFFFu) | ((tile_d0 & 0xFFFFu) << 16));
  g1[4] = (int)(tile_d1 & 0xFFFFu);                           // tile_dim1, tile_dim2=0
  g1[5] = (int)stride0;                                       // dim0 stride lo
  g1[6] = 0;                                                  // stride hi / dim1 stride lo
  g1[7] = 0;
  i32x4 z4 = {0, 0, 0, 0};
  i32x8 z8 = {0, 0, 0, 0, 0, 0, 0, 0};
  __builtin_amdgcn_tensor_load_to_lds(g0, g1, z4, z4, z8, 0);
}

// Load one 16x32 bf16 WMMA operand fragment (A as MxK row-major, or B as NxK
// row-major). Lane L holds row (L&15) at k=[koff..koff+7],[koff+16..koff+23],
// koff=(L>>4)*8 -> two 16-byte contiguous loads. Works on global or LDS ptrs.
static __device__ __forceinline__ v16bf load_frag(const __bf16* src, int ld) {
  const int lane = threadIdx.x & 31;
  const int row  = lane & 15;
  const int koff = (lane >> 4) << 3;
  const __bf16* p = src + (size_t)row * ld + koff;
  v8bf a0 = *(const v8bf*)(p);
  v8bf a1 = *(const v8bf*)(p + 16);
  return __builtin_shufflevector(a0, a1, 0,1,2,3,4,5,6,7,8,9,10,11,12,13,14,15);
}

static __device__ __forceinline__ v8f wmma_bf16(v16bf a, v16bf b, v8f c) {
  return __builtin_amdgcn_wmma_f32_16x16x32_bf16(false, a, false, b,
                                                 (short)0, c, false, false);
}

// ---------------------------------------------------------------- convert
__global__ void cvt_f32_bf16_kernel(const float* __restrict__ src,
                                    void* __restrict__ dstv, int n4) {
  __bf16* dst = (__bf16*)dstv;
  int i = blockIdx.x * blockDim.x + threadIdx.x;
  const int stride = gridDim.x * blockDim.x;
  for (; i < n4; i += stride) {
    float4 f = ((const float4*)src)[i];
    v4bf o;
    o.x = f2bf(f.x); o.y = f2bf(f.y); o.z = f2bf(f.z); o.w = f2bf(f.w);
    ((v4bf*)dst)[i] = o;
  }
}

// --------------------------------------------------- QKV projection + RoPE
// grid: (BS/64, H+KV+KV); block: 128 (4 waves); wave -> 16 rows x 128 cols.
// Weight tile (128 n x 32 k) is TDM-staged into LDS, double buffered; each
// wave DMAs its 32-row quarter.
__global__ void qkv_rope_kernel(const void* __restrict__ xbv,
                                const void* __restrict__ wqv,
                                const void* __restrict__ wkv,
                                const void* __restrict__ wvv,
                                const float* __restrict__ cosp,
                                const float* __restrict__ sinp,
                                void* __restrict__ qbv,
                                void* __restrict__ kbv,
                                void* __restrict__ vtv) {
  __shared__ __align__(16) __bf16 wtile[2][128 * 32];   // 2 x 8 KiB
  const __bf16* Xb = (const __bf16*)xbv;
  const int wave = threadIdx.x >> 5;
  const int lane = threadIdx.x & 31;
  const int half = lane >> 4;
  const int nlo  = lane & 15;
  const int rb   = blockIdx.x * 4 + wave;        // 16-row block id
  const int strip = blockIdx.y;                  // 0..23 (Q heads, K, V)

  const __bf16* W;
  int n0;
  if (strip < HNUM)              { W = (const __bf16*)wqv; n0 = strip * HD; }
  else if (strip < HNUM + KVNUM) { W = (const __bf16*)wkv; n0 = (strip - HNUM) * HD; }
  else                           { W = (const __bf16*)wvv; n0 = (strip - HNUM - KVNUM) * HD; }

  const unsigned wt0 = lds_off(&wtile[0][0]);
  const unsigned wt1 = lds_off(&wtile[1][0]);
  const __bf16* A = Xb + (size_t)rb * 16 * DDIM;
  v8f acc[8] = {};

  const int NK = DDIM / 32;
  // prologue: stage k-chunk 0
  tdm_load_2d(wt0 + wave * 2048, W + (size_t)(n0 + wave * 32) * DDIM, 32, 32, DDIM);
  for (int i = 0; i < NK; ++i) {
    const int kk = i * 32;
    if (i + 1 < NK) {
      const unsigned dst = ((i + 1) & 1) ? wt1 : wt0;
      tdm_load_2d(dst + wave * 2048,
                  W + (size_t)(n0 + wave * 32) * DDIM + (kk + 32), 32, 32, DDIM);
      __builtin_amdgcn_s_wait_tensorcnt(1);
    } else {
      __builtin_amdgcn_s_wait_tensorcnt(0);
    }
    __syncthreads();
    v16bf af = load_frag(A + kk, DDIM);
    const __bf16* bsrc = &wtile[i & 1][0];
#pragma unroll
    for (int t = 0; t < 8; ++t) {
      v16bf bf = load_frag(bsrc + t * 16 * 32, 32);
      acc[t] = wmma_bf16(af, bf, acc[t]);
    }
    __syncthreads();   // all reads done before buffer is overwritten
  }

  if (strip < HNUM + KVNUM) {
    // RoPE epilogue: column j pairs with j^64 -> tiles t and t+4, same lane.
    __bf16* Qb = (__bf16*)qbv;
    __bf16* Kb = (__bf16*)kbv;
#pragma unroll
    for (int v = 0; v < 8; ++v) {
      const int r = rb * 16 + v + half * 8;      // global token row (b*S+s)
      const int b = r >> 11;
      const int s = r & (SDIM - 1);
#pragma unroll
      for (int t = 0; t < 8; ++t) {
        const int j = t * 16 + nlo;
        const float c  = cosp[(size_t)r * HD + j];
        const float sn = sinp[(size_t)r * HD + j];
        const float x  = acc[t][v];
        const float y  = (t < 4) ? -acc[t + 4][v] : acc[t - 4][v];
        float o = x * c + y * sn;
        if (strip < HNUM) {
          o *= SCALEV;                           // fold score scale into Q
          Qb[(((size_t)b * HNUM + strip) * SDIM + s) * HD + j] = f2bf(o);
        } else {
          const int kv = strip - HNUM;
          Kb[(((size_t)b * KVNUM + kv) * SDIM + s) * HD + j] = f2bf(o);
        }
      }
    }
  } else {
    // V stored transposed [b][kv][hd][S]: P*V B-fragments read contiguous keys
    __bf16* Vt = (__bf16*)vtv;
    const int kv = strip - HNUM - KVNUM;
#pragma unroll
    for (int v = 0; v < 8; ++v) {
      const int r = rb * 16 + v + half * 8;
      const int b = r >> 11;
      const int s = r & (SDIM - 1);
#pragma unroll
      for (int t = 0; t < 8; ++t) {
        const int j = t * 16 + nlo;
        Vt[(((size_t)b * KVNUM + kv) * HD + j) * SDIM + s] = f2bf(acc[t][v]);
      }
    }
  }
}

// ------------------------------------------------------- flash attention
// grid: (S/64, H, B); block 128 (4 waves); wave -> 16 q rows, full head.
// K (32 keys x 128d) and Vt (128d x 32 keys) tiles are TDM-staged, double
// buffered: waves 0-1 DMA K halves, waves 2-3 DMA Vt halves.
__global__ void flash_attn_kernel(const void* __restrict__ qbv,
                                  const void* __restrict__ kbv,
                                  const void* __restrict__ vtv,
                                  void* __restrict__ obv) {
  __shared__ __align__(16) __bf16 ktile[2][32 * 128];   // 2 x 8 KiB
  __shared__ __align__(16) __bf16 vtile[2][128 * 32];   // 2 x 8 KiB
  __shared__ __align__(16) __bf16 pbuf[4][16 * 32];     // 4 KiB
  const __bf16* Qb = (const __bf16*)qbv;
  const __bf16* Kb = (const __bf16*)kbv;
  const __bf16* Vt = (const __bf16*)vtv;
  __bf16* Ob = (__bf16*)obv;

  const int wave = threadIdx.x >> 5;
  const int lane = threadIdx.x & 31;
  const int half = lane >> 4;
  const int nlo  = lane & 15;
  const int qtile = blockIdx.x * 4 + wave;
  const int h = blockIdx.y;
  const int b = blockIdx.z;
  const int kv = h / (HNUM / KVNUM);
  const int qbase = qtile * 16;

  const __bf16* Qp = Qb + (((size_t)b * HNUM + h) * SDIM + qbase) * HD;
  const __bf16* Kp = Kb + ((size_t)b * KVNUM + kv) * SDIM * HD;
  const __bf16* Vp = Vt + ((size_t)b * KVNUM + kv) * HD * SDIM;

  const unsigned kt0 = lds_off(&ktile[0][0]);
  const unsigned vt0 = lds_off(&vtile[0][0]);

  v16bf qf[4];
#pragma unroll
  for (int c = 0; c < 4; ++c) qf[c] = load_frag(Qp + c * 32, HD);

  v8f o[8] = {};
  float mrow[8], lrow[8];
#pragma unroll
  for (int v = 0; v < 8; ++v) { mrow[v] = -1e30f; lrow[v] = 0.0f; }

  // Uniform trip count across all 4 waves (mask kills overshoot) so barriers
  // are convergent.
  const int kbmax = ((blockIdx.x * 4 + 3) * 16 + 15) / 32;

  // prologue: stage key-block 0
  if (wave < 2)
    tdm_load_2d(kt0 + wave * (16 * 128 * 2),
                Kp + (size_t)(wave * 16) * HD, 128, 16, HD);
  else
    tdm_load_2d(vt0 + (wave - 2) * (64 * 32 * 2),
                Vp + (size_t)((wave - 2) * 64) * SDIM, 32, 64, SDIM);

  for (int kb = 0; kb <= kbmax; ++kb) {
    if (kb + 1 <= kbmax) {
      const unsigned bsel = (unsigned)((kb + 1) & 1) * (32 * 128 * 2);
      if (wave < 2)
        tdm_load_2d(kt0 + bsel + wave * (16 * 128 * 2),
                    Kp + (size_t)((kb + 1) * 32 + wave * 16) * HD, 128, 16, HD);
      else
        tdm_load_2d(vt0 + bsel + (wave - 2) * (64 * 32 * 2),
                    Vp + (size_t)((wave - 2) * 64) * SDIM + (kb + 1) * 32, 32, 64, SDIM);
      __builtin_amdgcn_s_wait_tensorcnt(1);
    } else {
      __builtin_amdgcn_s_wait_tensorcnt(0);
    }
    __syncthreads();
    const __bf16* ks = &ktile[kb & 1][0];
    const __bf16* vs = &vtile[kb & 1][0];

    // S = Q * K^T for 32 keys (two 16-key score tiles), K from LDS
    v8f sc[2] = {};
#pragma unroll
    for (int c = 0; c < 2; ++c) {
#pragma unroll
      for (int kc = 0; kc < 4; ++kc) {
        v16bf kf = load_frag(ks + (size_t)(c * 16) * 128 + kc * 32, 128);
        sc[c] = wmma_bf16(qf[kc], kf, sc[c]);
      }
    }
    // tanh softcap + causal mask
#pragma unroll
    for (int c = 0; c < 2; ++c) {
      const int key = kb * 32 + c * 16 + nlo;
#pragma unroll
      for (int v = 0; v < 8; ++v) {
        const int row = qbase + v + half * 8;
        float s = sc[c][v];
        s = CAPV * tanhf(s * (1.0f / CAPV));
        sc[c][v] = (key <= row) ? s : -1e30f;
      }
    }
    // online softmax; row (v, lane-half) lives across a 16-lane group
#pragma unroll
    for (int v = 0; v < 8; ++v) {
      float mx = fmaxf(sc[0][v], sc[1][v]);
      mx = fmaxf(mx, __shfl_xor(mx, 1));
      mx = fmaxf(mx, __shfl_xor(mx, 2));
      mx = fmaxf(mx, __shfl_xor(mx, 4));
      mx = fmaxf(mx, __shfl_xor(mx, 8));
      const float mnew  = fmaxf(mrow[v], mx);
      const float alpha = __expf(mrow[v] - mnew);
      const float p0 = __expf(sc[0][v] - mnew);
      const float p1 = __expf(sc[1][v] - mnew);
      sc[0][v] = p0;
      sc[1][v] = p1;
      float ps = p0 + p1;
      ps += __shfl_xor(ps, 1);
      ps += __shfl_xor(ps, 2);
      ps += __shfl_xor(ps, 4);
      ps += __shfl_xor(ps, 8);
      lrow[v] = lrow[v] * alpha + ps;
      mrow[v] = mnew;
#pragma unroll
      for (int t = 0; t < 8; ++t) o[t][v] *= alpha;
    }
    // stage P through LDS: C-layout -> A-fragment layout
#pragma unroll
    for (int c = 0; c < 2; ++c)
#pragma unroll
      for (int v = 0; v < 8; ++v) {
        const int m = v + half * 8;
        pbuf[wave][m * 32 + c * 16 + nlo] = f2bf(sc[c][v]);
      }
    __syncthreads();
    v16bf pf = load_frag(&pbuf[wave][0], 32);
    // O += P * V  (Vt tile from LDS: contiguous keys per lane)
#pragma unroll
    for (int t = 0; t < 8; ++t) {
      v16bf vf = load_frag(vs + (size_t)(t * 16) * 32, 32);
      o[t] = wmma_bf16(pf, vf, o[t]);
    }
    __syncthreads();   // all reads done before buffers are overwritten
  }

  // normalize + store bf16 [b][s][h*HD+j]
#pragma unroll
  for (int v = 0; v < 8; ++v) {
    const float inv = 1.0f / lrow[v];
    const int s = qbase + v + half * 8;
#pragma unroll
    for (int t = 0; t < 8; ++t) {
      const int j = t * 16 + nlo;
      Ob[((size_t)b * SDIM + s) * (HNUM * HD) + h * HD + j] = f2bf(o[t][v] * inv);
    }
  }
}

// ------------------------------------------------------ output projection
__global__ void oproj_kernel(const void* __restrict__ obv,
                             const void* __restrict__ wov,
                             float* __restrict__ Out) {
  __shared__ __align__(16) __bf16 wtile[2][128 * 32];
  const __bf16* Ob  = (const __bf16*)obv;
  const __bf16* Wob = (const __bf16*)wov;
  const int wave = threadIdx.x >> 5;
  const int lane = threadIdx.x & 31;
  const int half = lane >> 4;
  const int nlo  = lane & 15;
  const int rb = blockIdx.x * 4 + wave;
  const int n0 = blockIdx.y * HD;

  const unsigned wt0 = lds_off(&wtile[0][0]);
  const unsigned wt1 = lds_off(&wtile[1][0]);
  const __bf16* A = Ob + (size_t)rb * 16 * DDIM;
  v8f acc[8] = {};

  const int NK = DDIM / 32;
  tdm_load_2d(wt0 + wave * 2048, Wob + (size_t)(n0 + wave * 32) * DDIM, 32, 32, DDIM);
  for (int i = 0; i < NK; ++i) {
    const int kk = i * 32;
    if (i + 1 < NK) {
      const unsigned dst = ((i + 1) & 1) ? wt1 : wt0;
      tdm_load_2d(dst + wave * 2048,
                  Wob + (size_t)(n0 + wave * 32) * DDIM + (kk + 32), 32, 32, DDIM);
      __builtin_amdgcn_s_wait_tensorcnt(1);
    } else {
      __builtin_amdgcn_s_wait_tensorcnt(0);
    }
    __syncthreads();
    v16bf af = load_frag(A + kk, DDIM);
    const __bf16* bsrc = &wtile[i & 1][0];
#pragma unroll
    for (int t = 0; t < 8; ++t) {
      v16bf bf = load_frag(bsrc + t * 16 * 32, 32);
      acc[t] = wmma_bf16(af, bf, acc[t]);
    }
    __syncthreads();
  }
#pragma unroll
  for (int v = 0; v < 8; ++v) {
    const int r = rb * 16 + v + half * 8;
#pragma unroll
    for (int t = 0; t < 8; ++t)
      Out[(size_t)r * DDIM + n0 + t * 16 + nlo] = acc[t][v];
  }
}

// ---------------------------------------------------------------- launch
extern "C" void kernel_launch(void* const* d_in, const int* in_sizes, int n_in,
                              void* d_out, int out_size, void* d_ws, size_t ws_size,
                              hipStream_t stream) {
  (void)in_sizes; (void)n_in; (void)out_size; (void)ws_size;
  const float* hs   = (const float*)d_in[0];
  const float* cosp = (const float*)d_in[1];
  const float* sinp = (const float*)d_in[2];
  const float* Wq   = (const float*)d_in[3];
  const float* Wk   = (const float*)d_in[4];
  const float* Wv   = (const float*)d_in[5];
  const float* Wo   = (const float*)d_in[6];
  float* out = (float*)d_out;
  char* ws = (char*)d_ws;

  // workspace layout (bytes)
  void* Xb  = ws;                       // 4096*2048  bf16 = 16 MiB
  void* Wqb = ws + (size_t)16777216;    //  8 MiB
  void* Wkb = ws + (size_t)25165824;    //  2 MiB
  void* Wvb = ws + (size_t)27262976;    //  2 MiB
  void* Wob = ws + (size_t)29360128;    //  8 MiB
  void* Qb  = ws + (size_t)37748736;    // 16 MiB
  void* Kb  = ws + (size_t)54525952;    //  4 MiB
  void* Vt  = ws + (size_t)58720256;    //  4 MiB (transposed V)
  void* Ob  = ws + (size_t)62914560;    // 16 MiB

  cvt_f32_bf16_kernel<<<2048, 256, 0, stream>>>(hs, Xb, BDIM * SDIM * DDIM / 4);
  cvt_f32_bf16_kernel<<<1024, 256, 0, stream>>>(Wq, Wqb, HNUM * HD * DDIM / 4);
  cvt_f32_bf16_kernel<<<256, 256, 0, stream>>>(Wk, Wkb, KVNUM * HD * DDIM / 4);
  cvt_f32_bf16_kernel<<<256, 256, 0, stream>>>(Wv, Wvb, KVNUM * HD * DDIM / 4);
  cvt_f32_bf16_kernel<<<1024, 256, 0, stream>>>(Wo, Wob, DDIM * HNUM * HD / 4);

  qkv_rope_kernel<<<dim3(64, HNUM + 2 * KVNUM), 128, 0, stream>>>(
      Xb, Wqb, Wkb, Wvb, cosp, sinp, Qb, Kb, Vt);

  flash_attn_kernel<<<dim3(SDIM / 64, HNUM, BDIM), 128, 0, stream>>>(Qb, Kb, Vt, Ob);

  oproj_kernel<<<dim3(64, DDIM / HD), 128, 0, stream>>>(Ob, Wob, out);
}
